// Attention_9732395893237
// MI455X (gfx1250) — matrix operand order repaired and checked by reading, not compiled, not run on hardware
//
#include <hip/hip_runtime.h>

#define B_     4
#define N_     1024
#define D_     1024
#define H_     16
#define DH_    64
#define INNER_ 1024
#define QKVC   3072

// LDS strides / tile sizes in u16 units. Strides chosen so async B128 chunks land
// 16B-aligned and 16-lane fragment reads are bank-conflict-free.
#define AS_STRIDE 40
#define AS_TILE   (64 * AS_STRIDE)
#define BT_STRIDE 36
#define BT_TILE   (64 * BT_STRIDE)
#define QS_STRIDE 72
#define KS_STRIDE 72
#define KS_TILE   (32 * KS_STRIDE)
#define VT_STRIDE 36
#define VT_TILE   (64 * VT_STRIDE)
#define PW_STRIDE 36

typedef __attribute__((ext_vector_type(16))) __bf16        v16bf;
typedef __attribute__((ext_vector_type(8)))  float         v8f;
typedef __attribute__((ext_vector_type(8)))  unsigned int  v8u;
typedef __attribute__((address_space(3))) const char       lds_cchar;

__device__ __forceinline__ unsigned short f2bf(float f) {
    unsigned u = __float_as_uint(f);
    u += 0x7FFFu + ((u >> 16) & 1u);        // round-to-nearest-even
    return (unsigned short)(u >> 16);
}

__device__ __forceinline__ float rmax16(float x) {
#pragma unroll
    for (int m = 1; m < 16; m <<= 1) x = fmaxf(x, __shfl_xor(x, m, 32));
    return x;
}
__device__ __forceinline__ float rsum16(float x) {
#pragma unroll
    for (int m = 1; m < 16; m <<= 1) x += __shfl_xor(x, m, 32);
    return x;
}

// 32-bit LDS byte offset of a shared-memory object (for async-to-LDS VDST operand)
__device__ __forceinline__ unsigned lds_off(const void* p) {
    return (unsigned)(size_t)(lds_cchar*)p;
}

// CDNA5 async copy: 16 bytes per lane, global -> LDS, tracked by ASYNCcnt (GVS mode)
__device__ __forceinline__ void async_ld_b128(const void* base, unsigned goff_bytes,
                                              unsigned lds_bytes) {
    asm volatile("global_load_async_to_lds_b128 %0, %1, %2"
                 :: "v"(lds_bytes), "v"(goff_bytes), "s"(base)
                 : "memory");
}
__device__ __forceinline__ void wait_async0() {
    asm volatile("s_wait_asynccnt 0x0" ::: "memory");
}

// A fragment 16x32 bf16 (ISA §7.12.2): lane m=L%16, half=L/16;
// VGPR v holds K = (v%4)*2 + half*8 + (v/4)*16 (+1). buf is m-major.
__device__ __forceinline__ v16bf load_frag_a(const unsigned short* buf, int stride,
                                             int rowBase, int kBase, int lane) {
    const int m = lane & 15, half = lane >> 4;
    v8u r;
#pragma unroll
    for (int v = 0; v < 8; ++v) {
        const int k = ((v & 3) << 1) + (half << 3) + ((v >> 2) << 4);
        r[v] = *(const unsigned int*)(buf + (rowBase + m) * stride + kBase + k);
    }
    return __builtin_bit_cast(v16bf, r);
}

// B fragment 32x16 bf16 from an n-major (k-minor) buffer buf[n][k]:
// lane n=L%16, half=L/16; VGPR v holds K = (v%4)*2 + half*8 + (v/4)*16 (+1).
__device__ __forceinline__ v16bf load_frag_b(const unsigned short* buf, int stride,
                                             int colBase, int kBase, int lane) {
    const int n = lane & 15, half = lane >> 4;
    v8u r;
#pragma unroll
    for (int v = 0; v < 8; ++v) {
        const int k = ((v & 3) << 1) + (half << 3) + ((v >> 2) << 4);
        r[v] = *(const unsigned int*)(buf + (colBase + n) * stride + kBase + k);
    }
    return __builtin_bit_cast(v16bf, r);
}

__global__ void k_cvt(const float* __restrict__ src, unsigned short* __restrict__ dst, int n) {
    int i = blockIdx.x * blockDim.x + threadIdx.x;
    const int stride = gridDim.x * blockDim.x;
    for (; i < n; i += stride) dst[i] = f2bf(src[i]);
}

// ---------------- QKV projection: qkv[4096][3072] = x[4096][1024] @ w[1024][3072]
__global__ __launch_bounds__(128) void k_gemm_qkv(const unsigned short* __restrict__ xb,
                                                  const unsigned short* __restrict__ wb,
                                                  unsigned short* __restrict__ qkvb) {
    __shared__ unsigned short As[2 * AS_TILE];   // [row][k], double buffered
    __shared__ unsigned short Bt[2 * BT_TILE];   // [n][k] transposed, double buffered
    const int t = threadIdx.x, lane = t & 31, wid = t >> 5;
    const int n0 = blockIdx.x * 64, m0 = blockIdx.y * 64;

    auto load_a = [&](int bufi, int k0) {        // 64x32 tile, async 16B/lane chunks
#pragma unroll
        for (int i = 0; i < 2; ++i) {
            const int idx = t + 128 * i;
            const int r = idx >> 2, seg = idx & 3;
            const unsigned go = (unsigned)(((m0 + r) * D_ + k0 + seg * 8) * 2);
            async_ld_b128(xb, go, lds_off(As + bufi * AS_TILE + r * AS_STRIDE + seg * 8));
        }
    };
    auto load_b = [&](int bufi, int k0) {        // 32x64 tile, register transpose
#pragma unroll
        for (int i = 0; i < 2; ++i) {
            const int idx = t + 128 * i;
            const int kr = idx >> 3, seg = idx & 7;
            const uint4 dv = *reinterpret_cast<const uint4*>(wb + (size_t)(k0 + kr) * QKVC + n0 + seg * 8);
            unsigned int w4[4] = {dv.x, dv.y, dv.z, dv.w};
#pragma unroll
            for (int e = 0; e < 8; ++e)
                Bt[bufi * BT_TILE + (seg * 8 + e) * BT_STRIDE + kr] =
                    (unsigned short)(w4[e >> 1] >> ((e & 1) * 16));
        }
    };

    v8f acc[4] = {};
    load_a(0, 0);
    load_b(0, 0);
    wait_async0();
    __syncthreads();
    int buf = 0;
    for (int k0 = 0; k0 < D_; k0 += 32, buf ^= 1) {
        if (k0 + 32 < D_) { load_a(buf ^ 1, k0 + 32); load_b(buf ^ 1, k0 + 32); }
        const v16bf a = load_frag_a(As + buf * AS_TILE, AS_STRIDE, wid * 16, 0, lane);
#pragma unroll
        for (int nt = 0; nt < 4; ++nt) {
            const v16bf bfr = load_frag_b(Bt + buf * BT_TILE, BT_STRIDE, nt * 16, 0, lane);
            acc[nt] = __builtin_amdgcn_wmma_f32_16x16x32_bf16(false, a, false, bfr,
                                                              (short)0, acc[nt], false, false);
        }
        wait_async0();
        __syncthreads();
    }
    const int nlane = lane & 15, half = lane >> 4;
#pragma unroll
    for (int nt = 0; nt < 4; ++nt)
#pragma unroll
        for (int v = 0; v < 8; ++v) {
            const int row = m0 + wid * 16 + v + 8 * half;
            const int col = n0 + nt * 16 + nlane;
            qkvb[(size_t)row * QKVC + col] = f2bf(acc[nt][v]);
        }
}

// ---------------- Flash attention: 64 q rows of one (b,h) per block; ctx bf16 [4096][1024]
__global__ __launch_bounds__(128) void k_attn(const unsigned short* __restrict__ qkvb,
                                              const float* __restrict__ mask,
                                              unsigned short* __restrict__ ctx) {
    __shared__ unsigned short Qs[64 * QS_STRIDE];    // [qrow][d]
    __shared__ unsigned short Ks[2 * KS_TILE];       // [key][d], double buffered
    __shared__ unsigned short VT[2 * VT_TILE];       // [d][key], double buffered
    __shared__ unsigned short Pw[4][16 * PW_STRIDE]; // per-wave P tile [m][k]
    __shared__ float mqs[64];
    const int t = threadIdx.x, lane = t & 31, wid = t >> 5;
    const int blk = blockIdx.x;
    const int qt = blk & 15, h = (blk >> 4) & 15, b = blk >> 8;
    const int qrow0 = qt * 64;
    const int qcol = h * DH_;
    const int kcol = INNER_ + h * DH_;
    const int vcol = 2 * INNER_ + h * DH_;

    auto load_k = [&](int bufi, int jj) {            // 32x64, async
#pragma unroll
        for (int i = 0; i < 2; ++i) {
            const int idx = t + 128 * i;
            const int kr = idx >> 3, seg = idx & 7;
            const unsigned go = (unsigned)(((b * N_ + jj + kr) * QKVC + kcol + seg * 8) * 2);
            async_ld_b128(qkvb, go, lds_off(Ks + bufi * KS_TILE + kr * KS_STRIDE + seg * 8));
        }
    };
    auto load_v = [&](int bufi, int jj) {            // 32x64 -> VT[d][key] register transpose
#pragma unroll
        for (int i = 0; i < 2; ++i) {
            const int idx = t + 128 * i;
            const int kr = idx >> 3, seg = idx & 7;
            const uint4 dv = *reinterpret_cast<const uint4*>(
                qkvb + (size_t)(b * N_ + jj + kr) * QKVC + vcol + seg * 8);
            unsigned int w4[4] = {dv.x, dv.y, dv.z, dv.w};
#pragma unroll
            for (int e = 0; e < 8; ++e)
                VT[bufi * VT_TILE + (seg * 8 + e) * VT_STRIDE + kr] =
                    (unsigned short)(w4[e >> 1] >> ((e & 1) * 16));
        }
    };

#pragma unroll
    for (int i = 0; i < 4; ++i) {                    // Q tile 64x64, async
        const int idx = t + 128 * i;
        const int r = idx >> 3, seg = idx & 7;
        const unsigned go = (unsigned)(((b * N_ + qrow0 + r) * QKVC + qcol + seg * 8) * 2);
        async_ld_b128(qkvb, go, lds_off(Qs + r * QS_STRIDE + seg * 8));
    }
    if (t < 64) mqs[t] = mask[b * N_ + qrow0 + t];
    load_k(0, 0);
    load_v(0, 0);
    wait_async0();
    __syncthreads();

    const int nlane = lane & 15, half = lane >> 4;
    const v16bf qa0 = load_frag_a(Qs, QS_STRIDE, wid * 16, 0, lane);
    const v16bf qa1 = load_frag_a(Qs, QS_STRIDE, wid * 16, 32, lane);
    float mqv[8];
#pragma unroll
    for (int v = 0; v < 8; ++v) mqv[v] = mqs[wid * 16 + v + 8 * half];
    float mrun[8], lrun[8];
#pragma unroll
    for (int v = 0; v < 8; ++v) { mrun[v] = -__builtin_inff(); lrun[v] = 0.f; }
    v8f o[4] = {};
    const float scale = 0.125f;                      // DH^-0.5

    int buf = 0;
    for (int jj = 0; jj < N_; jj += 32, buf ^= 1) {
        if (jj + 32 < N_) { load_k(buf ^ 1, jj + 32); load_v(buf ^ 1, jj + 32); }

        float sarr[2][8];
#pragma unroll
        for (int st = 0; st < 2; ++st) {             // S = Q @ K^T
            const v16bf bk0 = load_frag_b(Ks + buf * KS_TILE, KS_STRIDE, st * 16, 0, lane);
            const v16bf bk1 = load_frag_b(Ks + buf * KS_TILE, KS_STRIDE, st * 16, 32, lane);
            v8f c = {};
            c = __builtin_amdgcn_wmma_f32_16x16x32_bf16(false, qa0, false, bk0, (short)0, c, false, false);
            c = __builtin_amdgcn_wmma_f32_16x16x32_bf16(false, qa1, false, bk1, (short)0, c, false, false);
            const float mk = mask[b * N_ + jj + st * 16 + nlane];
#pragma unroll
            for (int v = 0; v < 8; ++v) {
                const float val = c[v] * scale;
                sarr[st][v] = (mqv[v] * mk == 1.0f) ? val : -__builtin_inff();
            }
        }
        // online softmax per row (a row's 16 cols live across one half-wave in one VGPR)
#pragma unroll
        for (int v = 0; v < 8; ++v) {
            const float tmax = rmax16(fmaxf(sarr[0][v], sarr[1][v]));
            const float mnew = fmaxf(mrun[v], tmax);
            float alpha, p0, p1;
            if (mnew == -__builtin_inff()) { alpha = 1.f; p0 = 0.f; p1 = 0.f; }
            else {
                alpha = __expf(mrun[v] - mnew);
                p0 = __expf(sarr[0][v] - mnew);
                p1 = __expf(sarr[1][v] - mnew);
            }
            const float rs = rsum16(p0 + p1);
            lrun[v] = lrun[v] * alpha + rs;
            mrun[v] = mnew;
#pragma unroll
            for (int dt = 0; dt < 4; ++dt) o[dt][v] *= alpha;
            const int prow = (v + 8 * half) * PW_STRIDE;
            Pw[wid][prow + nlane]      = f2bf(p0);
            Pw[wid][prow + 16 + nlane] = f2bf(p1);
        }
        // O += P @ V
        const v16bf ap = load_frag_a(Pw[wid], PW_STRIDE, 0, 0, lane);
#pragma unroll
        for (int dt = 0; dt < 4; ++dt) {
            const v16bf bv = load_frag_b(VT + buf * VT_TILE, VT_STRIDE, dt * 16, 0, lane);
            o[dt] = __builtin_amdgcn_wmma_f32_16x16x32_bf16(false, ap, false, bv,
                                                            (short)0, o[dt], false, false);
        }
        wait_async0();
        __syncthreads();
    }
#pragma unroll
    for (int v = 0; v < 8; ++v) {
        const float inv = (lrun[v] > 0.f) ? 1.f / lrun[v] : 0.f;   // fully-masked row -> 0
        const int row = b * N_ + qrow0 + wid * 16 + v + 8 * half;
#pragma unroll
        for (int dt = 0; dt < 4; ++dt) {
            const int col = h * DH_ + dt * 16 + nlane;
            ctx[(size_t)row * INNER_ + col] = f2bf(o[dt][v] * inv);
        }
    }
}

// ---------------- Output projection: out[4096][1024] = ctx @ w_out + b_out (f32 out)
__global__ __launch_bounds__(128) void k_gemm_out(const unsigned short* __restrict__ ab,
                                                  const unsigned short* __restrict__ wb,
                                                  const float* __restrict__ bias,
                                                  float* __restrict__ out) {
    __shared__ unsigned short As[2 * AS_TILE];
    __shared__ unsigned short Bt[2 * BT_TILE];
    const int t = threadIdx.x, lane = t & 31, wid = t >> 5;
    const int n0 = blockIdx.x * 64, m0 = blockIdx.y * 64;

    auto load_a = [&](int bufi, int k0) {
#pragma unroll
        for (int i = 0; i < 2; ++i) {
            const int idx = t + 128 * i;
            const int r = idx >> 2, seg = idx & 3;
            const unsigned go = (unsigned)(((m0 + r) * INNER_ + k0 + seg * 8) * 2);
            async_ld_b128(ab, go, lds_off(As + bufi * AS_TILE + r * AS_STRIDE + seg * 8));
        }
    };
    auto load_b = [&](int bufi, int k0) {
#pragma unroll
        for (int i = 0; i < 2; ++i) {
            const int idx = t + 128 * i;
            const int kr = idx >> 3, seg = idx & 7;
            const uint4 dv = *reinterpret_cast<const uint4*>(wb + (size_t)(k0 + kr) * D_ + n0 + seg * 8);
            unsigned int w4[4] = {dv.x, dv.y, dv.z, dv.w};
#pragma unroll
            for (int e = 0; e < 8; ++e)
                Bt[bufi * BT_TILE + (seg * 8 + e) * BT_STRIDE + kr] =
                    (unsigned short)(w4[e >> 1] >> ((e & 1) * 16));
        }
    };

    v8f acc[4] = {};
    load_a(0, 0);
    load_b(0, 0);
    wait_async0();
    __syncthreads();
    int buf = 0;
    for (int k0 = 0; k0 < INNER_; k0 += 32, buf ^= 1) {
        if (k0 + 32 < INNER_) { load_a(buf ^ 1, k0 + 32); load_b(buf ^ 1, k0 + 32); }
        const v16bf a = load_frag_a(As + buf * AS_TILE, AS_STRIDE, wid * 16, 0, lane);
#pragma unroll
        for (int nt = 0; nt < 4; ++nt) {
            const v16bf bfr = load_frag_b(Bt + buf * BT_TILE, BT_STRIDE, nt * 16, 0, lane);
            acc[nt] = __builtin_amdgcn_wmma_f32_16x16x32_bf16(false, a, false, bfr,
                                                              (short)0, acc[nt], false, false);
        }
        wait_async0();
        __syncthreads();
    }
    const int nlane = lane & 15, half = lane >> 4;
#pragma unroll
    for (int nt = 0; nt < 4; ++nt) {
        const int col = n0 + nt * 16 + nlane;
        const float bv = bias[col];
#pragma unroll
        for (int v = 0; v < 8; ++v) {
            const int row = m0 + wid * 16 + v + 8 * half;
            out[(size_t)row * D_ + col] = acc[nt][v] + bv;
        }
    }
}

extern "C" void kernel_launch(void* const* d_in, const int* in_sizes, int n_in,
                              void* d_out, int out_size, void* d_ws, size_t ws_size,
                              hipStream_t stream) {
    (void)in_sizes; (void)n_in; (void)out_size; (void)ws_size;
    const float* x     = (const float*)d_in[0];
    const float* mask  = (const float*)d_in[1];
    const float* w_qkv = (const float*)d_in[2];
    const float* w_out = (const float*)d_in[3];
    const float* b_out = (const float*)d_in[4];
    float* out = (float*)d_out;

    unsigned short* ws = (unsigned short*)d_ws;
    const size_t n_x    = (size_t)B_ * N_ * D_;       // 4,194,304
    const size_t n_wqkv = (size_t)D_ * QKVC;          // 3,145,728
    const size_t n_wout = (size_t)INNER_ * D_;        // 1,048,576
    const size_t n_qkv  = (size_t)B_ * N_ * QKVC;     // 12,582,912
    unsigned short* xb    = ws;
    unsigned short* wqkvb = xb + n_x;
    unsigned short* woutb = wqkvb + n_wqkv;
    unsigned short* qkvb  = woutb + n_wout;
    unsigned short* ctx   = qkvb + n_qkv;             // total ~48 MB bf16 scratch

    k_cvt<<<2048, 256, 0, stream>>>(x, xb, (int)n_x);
    k_cvt<<<2048, 256, 0, stream>>>(w_qkv, wqkvb, (int)n_wqkv);
    k_cvt<<<1024, 256, 0, stream>>>(w_out, woutb, (int)n_wout);
    k_gemm_qkv<<<dim3(QKVC / 64, (B_ * N_) / 64), 128, 0, stream>>>(xb, wqkvb, qkvb);
    k_attn<<<B_ * H_ * (N_ / 64), 128, 0, stream>>>(qkvb, mask, ctx);
    k_gemm_out<<<dim3(D_ / 64, (B_ * N_) / 64), 128, 0, stream>>>(ctx, woutb, b_out, out);
}
